// CausalSelfAttention_16303695855972
// MI455X (gfx1250) — matrix operand order repaired and checked by reading, not compiled, not run on hardware
//
#include <hip/hip_runtime.h>
#include <cstdint>
#include <cstddef>

// ---------------------------------------------------------------------------
// CDNA5 (gfx1250) causal self-attention: bf16 WMMA GEMMs + flash attention.
// ---------------------------------------------------------------------------

typedef __bf16 bf16;
typedef __attribute__((ext_vector_type(16))) __bf16 v16bf;
typedef __attribute__((ext_vector_type(8)))  __bf16 v8bf;
typedef __attribute__((ext_vector_type(8)))  float  v8f;

constexpr int T  = 2048;
constexpr int C  = 4096;
constexpr int H  = 32;
constexpr int HS = 128;
constexpr int NE = 32;
constexpr int C3 = 3 * C;

// float -> bf16 (round half up; used on scalar paths)
__device__ __forceinline__ bf16 f2bf(float f) {
  uint32_t u = __builtin_bit_cast(uint32_t, f) + 0x8000u;
  uint16_t hu = (uint16_t)(u >> 16);
  return __builtin_bit_cast(bf16, hu);
}

// Pack two floats into two bf16 (round half up) with a single v_perm_b32.
__device__ __forceinline__ uint32_t pack2bf(float a, float b) {
  uint32_t ua = __builtin_bit_cast(uint32_t, a) + 0x8000u;
  uint32_t ub = __builtin_bit_cast(uint32_t, b) + 0x8000u;
  return __builtin_amdgcn_perm(ub, ua, 0x07060302u);
}

__device__ __forceinline__ v8f wmma_bf16(v16bf a, v16bf b, v8f c) {
  return __builtin_amdgcn_wmma_f32_16x16x32_bf16(false, a, false, b, (short)0, c,
                                                 false, false);
}

// Async DMA: 16 bytes per lane, global -> LDS, tracked on ASYNCcnt.
// ISA aperture rule: low 32 bits of a flat shared address are the LDS byte
// offset, so truncating the __shared__ pointer yields the VDST operand.
__device__ __forceinline__ void async_copy_b128(void* lds_dst, const void* gsrc) {
  const uint32_t dst = (uint32_t)(uintptr_t)lds_dst;
  asm volatile("global_load_async_to_lds_b128 %0, %1, off"
               :: "v"(dst), "v"(gsrc) : "memory");
}
__device__ __forceinline__ void wait_asynccnt0() {
  asm volatile("s_wait_asynccnt 0" ::: "memory");
}

// A-fragment (16x32 bf16, M x K): lane half kh holds K = kh*16+0..15; M = lane&15.
__device__ __forceinline__ v16bf load_a_frag(const bf16* p0, int row_base, int ldk,
                                             int k0, int lane) {
  const int m  = lane & 15;
  const int kh = lane >> 4;
  const bf16* p = p0 + (size_t)(row_base + m) * ldk + k0;
  return *(const v16bf*)(p + kh * 16);
}

// B-fragment (32x16 bf16, K x N): lanes 0-15 hold K=0..15, lanes 16-31 K=16..31;
// VGPR v packs K=2v,2v+1; N = lane&15.  Source must be [N][K] row-major.
__device__ __forceinline__ v16bf load_b_frag(const bf16* p0, int n_base, int ldk,
                                             int k0, int lane) {
  const int n  = lane & 15;
  const int kh = lane >> 4;
  return *(const v16bf*)(p0 + (size_t)(n_base + n) * ldk + k0 + kh * 16);
}

// ---------------------------------------------------------------------------
// GEMM:  Out[M][N] = A[M][K] @ W[N][K]^T + bias[N]    (bf16 WMMA, fp32 accum)
// Block tile 256x128, BK=32, double-buffered LDS (1 barrier / K-step).
// 8 waves in a 4x2 grid; each wave computes a 64x64 tile (4x4 WMMAs).
// Grid: blockIdx.x = M tile (fast) so concurrently-running blocks share the
// same 2MB weight tile through L2; the activation matrix is L2-resident.
// ---------------------------------------------------------------------------
template <typename AT>
__global__ __launch_bounds__(256) void gemm_bias_wmma(
    const AT* __restrict__ A, const float* __restrict__ W,
    const float* __restrict__ bias, float* __restrict__ Out,
    int M, int N, int K) {
  __shared__ bf16 As[2][256 * 32];   // [row][k]
  __shared__ bf16 Bs[2][128 * 32];   // [n][k]

  const int tid  = threadIdx.x;
  const int lane = tid & 31;
  const int wave = tid >> 5;
  const int wm   = wave & 3;    // 4 row groups x 64 rows
  const int wn   = wave >> 2;   // 2 col groups x 64 cols
  const int m0   = blockIdx.x * 256;
  const int n0   = blockIdx.y * 128;

  v8f acc[4][4] = {};

  // Staging: A -> one full 32-wide row slice per thread (row = tid);
  // B -> two threads per row (row = tid>>1, 16-elem halves).
  const int br   = tid >> 1;
  const int bseg = (tid & 1) * 16;

  float4 raf[8];   // fp32 A staging regs
  uint4  rab[2];   // bf16 A staging regs
  uint4  rab2[2];
  float4 rbf[4];   // B staging regs (fp32)

  auto load_tile = [&](int k0) {
    if constexpr (sizeof(AT) == 4) {
      const float4* sa4 = (const float4*)(A + (size_t)(m0 + tid) * K + k0);
#pragma unroll
      for (int i = 0; i < 8; ++i) raf[i] = sa4[i];
    } else {
      const uint4* sa4 = (const uint4*)(A + (size_t)(m0 + tid) * K + k0);
      rab[0] = sa4[0]; rab[1] = sa4[1]; rab2[0] = sa4[2]; rab2[1] = sa4[3];
    }
    const float4* sb4 = (const float4*)(W + (size_t)(n0 + br) * K + k0 + bseg);
#pragma unroll
    for (int i = 0; i < 4; ++i) rbf[i] = sb4[i];
  };

  auto store_tile = [&](int buf) {
    bf16* da = &As[buf][tid * 32];
    if constexpr (sizeof(AT) == 4) {
#pragma unroll
      for (int i = 0; i < 2; ++i) {
        uint4 w;
        w.x = pack2bf(raf[4 * i + 0].x, raf[4 * i + 0].y);
        w.y = pack2bf(raf[4 * i + 0].z, raf[4 * i + 0].w);
        w.z = pack2bf(raf[4 * i + 1].x, raf[4 * i + 1].y);
        w.w = pack2bf(raf[4 * i + 1].z, raf[4 * i + 1].w);
        uint4 w2;
        w2.x = pack2bf(raf[4 * i + 2].x, raf[4 * i + 2].y);
        w2.y = pack2bf(raf[4 * i + 2].z, raf[4 * i + 2].w);
        w2.z = pack2bf(raf[4 * i + 3].x, raf[4 * i + 3].y);
        w2.w = pack2bf(raf[4 * i + 3].z, raf[4 * i + 3].w);
        ((uint4*)da)[2 * i]     = w;
        ((uint4*)da)[2 * i + 1] = w2;
      }
    } else {
      ((uint4*)da)[0] = rab[0];  ((uint4*)da)[1] = rab[1];
      ((uint4*)da)[2] = rab2[0]; ((uint4*)da)[3] = rab2[1];
    }
    bf16* db = &Bs[buf][br * 32 + bseg];
    uint4 w;
    w.x = pack2bf(rbf[0].x, rbf[0].y); w.y = pack2bf(rbf[0].z, rbf[0].w);
    w.z = pack2bf(rbf[1].x, rbf[1].y); w.w = pack2bf(rbf[1].z, rbf[1].w);
    uint4 w2;
    w2.x = pack2bf(rbf[2].x, rbf[2].y); w2.y = pack2bf(rbf[2].z, rbf[2].w);
    w2.z = pack2bf(rbf[3].x, rbf[3].y); w2.w = pack2bf(rbf[3].z, rbf[3].w);
    ((uint4*)db)[0] = w; ((uint4*)db)[1] = w2;
  };

  load_tile(0);
  int cur = 0;
  for (int k0 = 0; k0 < K; k0 += 32) {
    store_tile(cur);
    __syncthreads();
    if (k0 + 32 < K) load_tile(k0 + 32);   // next tile in flight during WMMAs

    const bf16* Ab = &As[cur][0];
    const bf16* Bb = &Bs[cur][0];
    v16bf af[4];
#pragma unroll
    for (int tm = 0; tm < 4; ++tm)
      af[tm] = load_a_frag(Ab, wm * 64 + tm * 16, 32, 0, lane);
#pragma unroll
    for (int tn = 0; tn < 4; ++tn) {
      v16bf bfrag = load_b_frag(Bb, wn * 64 + tn * 16, 32, 0, lane);
#pragma unroll
      for (int tm = 0; tm < 4; ++tm)
        acc[tm][tn] = wmma_bf16(af[tm], bfrag, acc[tm][tn]);
    }
    cur ^= 1;
  }

  // C/D layout: VGPR j holds row (j + 8*(lane>>4)), col = lane&15.
  const int nlan = lane & 15;
  const int kh   = lane >> 4;
#pragma unroll
  for (int tm = 0; tm < 4; ++tm)
#pragma unroll
    for (int tn = 0; tn < 4; ++tn) {
      const int col = n0 + wn * 64 + tn * 16 + nlan;
      const float bv = bias[col];
#pragma unroll
      for (int j = 0; j < 8; ++j) {
        const int row = m0 + wm * 64 + tm * 16 + j + 8 * kh;
        Out[(size_t)row * N + col] = acc[tm][tn][j] + bv;
      }
    }
}

// ---------------------------------------------------------------------------
// RoPE + head split: qkv fp32 [T][H*3*HS] -> q,k,v bf16 [H][T][HS]
// ---------------------------------------------------------------------------
__global__ __launch_bounds__(256) void rope_split_kernel(
    const float* __restrict__ qkv, const float* __restrict__ cosb,
    const float* __restrict__ sinb, bf16* __restrict__ q,
    bf16* __restrict__ k, bf16* __restrict__ v) {
  const int idx = blockIdx.x * blockDim.x + threadIdx.x;   // over T*H*HS
  const int d = idx & (HS - 1);
  const int h = (idx >> 7) & (H - 1);
  const int t = idx >> 12;
  const float* base = qkv + (size_t)t * C3 + h * (3 * HS);
  float qv = base[d];
  float kv = base[HS + d];
  float vv = base[2 * HS + d];
  if (d < NE) {
    const float c = cosb[t * NE + d];
    const float s = sinb[t * NE + d];
    if (d < NE / 2) {       // rotated = -x2
      qv = base[d] * c - base[d + NE / 2] * s;
      kv = base[HS + d] * c - base[HS + d + NE / 2] * s;
    } else {                // rotated = +x1
      qv = base[d] * c + base[d - NE / 2] * s;
      kv = base[HS + d] * c + base[HS + d - NE / 2] * s;
    }
  }
  const size_t o = ((size_t)h * T + t) * HS + d;
  q[o] = f2bf(qv);
  k[o] = f2bf(kv);
  v[o] = f2bf(vv);
}

// ---------------------------------------------------------------------------
// Flash attention: per (128-query block, head) workgroup; 8 waves, each wave
// owns 16 query rows; KV processed in blocks of 64; online softmax in fp32.
// K tile staged with async global->LDS DMA (ASYNCcnt path).
// ---------------------------------------------------------------------------
__global__ __launch_bounds__(256) void flash_attn_kernel(
    const bf16* __restrict__ Q, const bf16* __restrict__ K,
    const bf16* __restrict__ V, bf16* __restrict__ Y) {
  __shared__ bf16 Ks [64 * 128];    // [key][d]   (B for Q@K^T: [n=key][k=d])
  __shared__ bf16 Vts[128 * 64];    // [d][key]   (B for P@V:   [n=d][k=key])
  __shared__ bf16 Ps [8][16 * 64];  // per-wave P tile, [row][key] (A for P@V)

  const int tid  = threadIdx.x;
  const int lane = tid & 31;
  const int wave = tid >> 5;
  const int h  = blockIdx.y;
  const int qb = blockIdx.x;
  const int q0 = qb * 128;

  const bf16* Qh = Q + (size_t)h * T * HS;
  const bf16* Kh = K + (size_t)h * T * HS;
  const bf16* Vh = V + (size_t)h * T * HS;

  // This wave's 16 query rows: hold all 4 K-step A-fragments in registers.
  v16bf qf[4];
#pragma unroll
  for (int ks = 0; ks < 4; ++ks)
    qf[ks] = load_a_frag(Qh + (size_t)(q0 + wave * 16) * HS, 0, HS, ks * 32, lane);

  v8f O[8] = {};
  float mrow[8], lrow[8];
#pragma unroll
  for (int j = 0; j < 8; ++j) { mrow[j] = -__builtin_inff(); lrow[j] = 0.f; }

  const int nlan = lane & 15;
  const int kh   = lane >> 4;
  const float scale = 0.08838834764831845f;  // 1/sqrt(HS)

  const int nkv = 2 * (qb + 1);
  for (int kb = 0; kb < nkv; ++kb) {
    const int kv0 = kb * 64;
    __syncthreads();  // all waves done reading previous K/V tiles
    // Stage K block (straight copy, [key][d]) via async global->LDS DMA:
    // 16B per lane per op, no VGPR round-trip, tracked on ASYNCcnt.
#pragma unroll
    for (int i = 0; i < 4; ++i) {
      const int it = tid + i * 256;               // 0..1023 16B chunks
      const int rr = it >> 4, cc = (it & 15) * 8;
      async_copy_b128(&Ks[rr * 128 + cc],
                      &Kh[(size_t)(kv0 + rr) * HS + cc]);
    }
    // Stage V transposed ([d][key]) so B-fragments are contiguous LDS reads
    for (int i = tid; i < 64 * 16; i += 256) {
      const int rr = i >> 4, cc = (i & 15) * 8;   // rr = key row, cc = d offset
      const bf16* src = Vh + (size_t)(kv0 + rr) * HS + cc;
#pragma unroll
      for (int u = 0; u < 8; ++u) Vts[(cc + u) * 64 + rr] = src[u];
    }
    wait_asynccnt0();   // K DMA complete before the barrier
    __syncthreads();

    // S = (Q @ K^T) for this wave's 16 rows x 64 keys (4 tiles)
    v8f S[4] = {};
#pragma unroll
    for (int ks = 0; ks < 4; ++ks) {
#pragma unroll
      for (int tn = 0; tn < 4; ++tn) {
        v16bf bfrag = load_b_frag(Ks, tn * 16, 128, ks * 32, lane);
        S[tn] = wmma_bf16(qf[ks], bfrag, S[tn]);
      }
    }

    // Scale + causal mask (C layout: row = j + 8*kh, col = nlan)
#pragma unroll
    for (int tn = 0; tn < 4; ++tn)
#pragma unroll
      for (int j = 0; j < 8; ++j) {
        float s = S[tn][j] * scale;
        const int kcol = kv0 + tn * 16 + nlan;
        const int qrow = q0 + wave * 16 + j + 8 * kh;
        if (kcol > qrow) s = -__builtin_inff();
        S[tn][j] = s;
      }

    // Online softmax: per-row reductions across the 16 lanes of a half-wave
    float alpha[8];
#pragma unroll
    for (int j = 0; j < 8; ++j) {
      float r = -__builtin_inff();
#pragma unroll
      for (int tn = 0; tn < 4; ++tn) r = fmaxf(r, S[tn][j]);
#pragma unroll
      for (int off = 1; off < 16; off <<= 1) r = fmaxf(r, __shfl_xor(r, off, 32));
      const float mnew = fmaxf(mrow[j], r);
      alpha[j] = __expf(mrow[j] - mnew);
      mrow[j]  = mnew;
      float rs = 0.f;
#pragma unroll
      for (int tn = 0; tn < 4; ++tn) {
        const float p = __expf(S[tn][j] - mnew);
        rs += p;
        Ps[wave][(j + 8 * kh) * 64 + tn * 16 + nlan] = f2bf(p);  // A-layout staging
      }
#pragma unroll
      for (int off = 1; off < 16; off <<= 1) rs += __shfl_xor(rs, off, 32);
      lrow[j] = lrow[j] * alpha[j] + rs;
    }

    // Rescale running output
#pragma unroll
    for (int dt = 0; dt < 8; ++dt)
#pragma unroll
      for (int j = 0; j < 8; ++j) O[dt][j] *= alpha[j];

    // Same-wave LDS RAW on Ps: DS ops are in-order, but wait out the counter.
    asm volatile("s_wait_dscnt 0" ::: "memory");

    // O += P @ V
#pragma unroll
    for (int ks = 0; ks < 2; ++ks) {
      const v16bf pf = load_a_frag(Ps[wave], 0, 64, ks * 32, lane);
#pragma unroll
      for (int dt = 0; dt < 8; ++dt) {
        v16bf bfrag = load_b_frag(Vts, dt * 16, 64, ks * 32, lane);
        O[dt] = wmma_bf16(pf, bfrag, O[dt]);
      }
    }
  }

  // Epilogue: y[t][h*HS + d] = O / l   (bf16, feeds the projection GEMM)
#pragma unroll
  for (int dt = 0; dt < 8; ++dt)
#pragma unroll
    for (int j = 0; j < 8; ++j) {
      const int tq = q0 + wave * 16 + j + 8 * kh;
      const int dc = dt * 16 + nlan;
      Y[(size_t)tq * C + h * HS + dc] = f2bf(O[dt][j] / lrow[j]);
    }
}

// ---------------------------------------------------------------------------
extern "C" void kernel_launch(void* const* d_in, const int* in_sizes, int n_in,
                              void* d_out, int out_size, void* d_ws, size_t ws_size,
                              hipStream_t stream) {
  (void)in_sizes; (void)n_in; (void)out_size; (void)ws_size;
  const float* x      = (const float*)d_in[0];
  const float* cosb   = (const float*)d_in[1];
  const float* sinb   = (const float*)d_in[2];
  const float* W_attn = (const float*)d_in[3];
  const float* b_attn = (const float*)d_in[4];
  const float* W_proj = (const float*)d_in[5];
  const float* b_proj = (const float*)d_in[6];
  float* out = (float*)d_out;

  // Workspace layout (160 MiB total):
  float* qkv = (float*)d_ws;                    // [T][3C] fp32   (96 MiB)
  bf16* q = (bf16*)(qkv + (size_t)T * C3);      // [H][T][HS]     (16 MiB)
  bf16* k = q + (size_t)H * T * HS;             // [H][T][HS]     (16 MiB)
  bf16* v = k + (size_t)H * T * HS;             // [H][T][HS]     (16 MiB)
  bf16* y = v + (size_t)H * T * HS;             // [T][C]         (16 MiB)

  // 1) qkv = x @ W_attn^T + b_attn    (M fast in grid -> weight-tile L2 reuse)
  gemm_bias_wmma<float><<<dim3(T / 256, C3 / 128), 256, 0, stream>>>(
      x, W_attn, b_attn, qkv, T, C3, C);
  // 2) RoPE + split into per-head bf16 q/k/v
  rope_split_kernel<<<(T * H * HS) / 256, 256, 0, stream>>>(qkv, cosb, sinb, q, k, v);
  // 3) causal flash attention
  flash_attn_kernel<<<dim3(T / 128, H), 256, 0, stream>>>(q, k, v, y);
  // 4) out = y @ W_proj^T + b_proj
  gemm_bias_wmma<bf16><<<dim3(T / 256, C / 128), 256, 0, stream>>>(
      y, W_proj, b_proj, out, T, C, C);
}
